// MeshCrossAttention_4973572128940
// MI455X (gfx1250) — compile-verified, matrix operand order checked by reading, not run on hardware
//
#include <hip/hip_runtime.h>

// ---------------- problem constants ----------------
#define Dn   1024
#define NH   16
#define HDn  64
#define NJ   3
#define NB   16
#define LQn  512
#define LKn  512

// ---------------- WMMA fragment types ----------------
typedef __attribute__((ext_vector_type(16))) __bf16          v16bf;
typedef __attribute__((ext_vector_type(16))) unsigned short  v16u;
typedef __attribute__((ext_vector_type(8)))  float           v8f;

// float -> bf16 (round to nearest even, bit math: always compiles)
__device__ __forceinline__ unsigned short f2bf(float f) {
    unsigned int u = __builtin_bit_cast(unsigned int, f);
    u += 0x7FFFu + ((u >> 16) & 1u);
    return (unsigned short)(u >> 16);
}

// One 16x16x32 bf16 WMMA, fp32 accumulate.
__device__ __forceinline__ v8f wmma_bf16(v16u a, v16u b, v8f c) {
    return __builtin_amdgcn_wmma_f32_16x16x32_bf16(
        false, __builtin_bit_cast(v16bf, a),
        false, __builtin_bit_cast(v16bf, b),
        (short)0, c, false, false);
}

// A/B operand fragment from row-major bf16 [rows x K], row stride ld (elems).
// CDNA5 16-bit layout: lanes 0-15 -> K = [0..7] & [16..23]; lanes 16-31 -> +8.
// Two 16B vector loads per lane (global_load_b128 / ds_load_b128).
__device__ __forceinline__ v16u ldfrag_bf(const unsigned short* base, int ld, int lane) {
    const int r  = lane & 15;
    const int kh = (lane >> 4) << 3;
    const unsigned short* p = base + (size_t)r * ld + kh;
    union { uint4 q[2]; v16u v; } u;
    u.q[0] = *reinterpret_cast<const uint4*>(p);
    u.q[1] = *reinterpret_cast<const uint4*>(p + 16);
    return u.v;
}

// ==================================================================
// Kernel 0: bulk fp32 -> bf16 cast (memory bound; keeps VALU out of GEMMs).
// 8 elements per thread, b128 in / b128 out. n must be a multiple of 8.
// ==================================================================
__global__ __launch_bounds__(256) void cast_f32_bf16_kernel(
    const float* __restrict__ in, unsigned short* __restrict__ out, size_t n)
{
    const size_t i = ((size_t)blockIdx.x * blockDim.x + threadIdx.x) * 8;
    if (i + 8 > n) return;
    const float4 a = *reinterpret_cast<const float4*>(in + i);
    const float4 b = *reinterpret_cast<const float4*>(in + i + 4);
    union { unsigned short s[8]; uint4 v; } o;
    o.s[0] = f2bf(a.x); o.s[1] = f2bf(a.y); o.s[2] = f2bf(a.z); o.s[3] = f2bf(a.w);
    o.s[4] = f2bf(b.x); o.s[5] = f2bf(b.y); o.s[6] = f2bf(b.z); o.s[7] = f2bf(b.w);
    *reinterpret_cast<uint4*>(out + i) = o.v;
}

// ==================================================================
// Kernel 1: projection GEMM  C(bf16, MxN) = A(bf16, MxK) @ W(bf16, NxK)^T + bias
// Block: 256 thr = 8 waves; block tile 128x128; wave tile 32x64 (2x4 wmma accs)
// Inner loop: pure b128 loads + v_wmma (no conversions).
// ==================================================================
__global__ __launch_bounds__(256) void proj_gemm_kernel(
    const unsigned short* __restrict__ A, const unsigned short* __restrict__ W,
    const float* __restrict__ bias, unsigned short* __restrict__ C,
    int M, int N, int K)
{
    const int lane  = threadIdx.x & 31;
    const int wid   = threadIdx.x >> 5;
    const int waveM = wid & 3;
    const int waveN = wid >> 2;
    const int m0 = blockIdx.y * 128 + waveM * 32;
    const int n0 = blockIdx.x * 128 + waveN * 64;

    v8f acc[2][4] = {};

    for (int k0 = 0; k0 < K; k0 += 32) {
        v16u a0 = ldfrag_bf(A + (size_t)m0 * K + k0, K, lane);
        v16u a1 = ldfrag_bf(A + (size_t)(m0 + 16) * K + k0, K, lane);
#pragma unroll
        for (int nt = 0; nt < 4; ++nt) {
            v16u bf = ldfrag_bf(W + (size_t)(n0 + nt * 16) * K + k0, K, lane);
            acc[0][nt] = wmma_bf16(a0, bf, acc[0][nt]);
            acc[1][nt] = wmma_bf16(a1, bf, acc[1][nt]);
        }
    }

    const int r16 = lane & 15, half = lane >> 4;
#pragma unroll
    for (int nt = 0; nt < 4; ++nt) {
        const int col = n0 + nt * 16 + r16;
        const float bv = bias[col];
#pragma unroll
        for (int mt = 0; mt < 2; ++mt)
#pragma unroll
            for (int i = 0; i < 8; ++i) {
                const int row = m0 + mt * 16 + half * 8 + i;
                C[(size_t)row * N + col] = f2bf(acc[mt][nt][i] + bv);
            }
    }
}

// ==================================================================
// Kernel 2: flash attention per (j,b,h).
// grid = (LQ/64, J*B*H), block = 128 thr (4 waves).
// Wave w owns 16 query rows; Q frags resident in VGPRs; stream K/V in 64-key
// tiles; online softmax; P via LDS (bf16); V transposed in LDS for B-operand.
// ==================================================================
__global__ __launch_bounds__(128) void attn_kernel(
    const unsigned short* __restrict__ Q,
    const unsigned short* __restrict__ Kk,
    const unsigned short* __restrict__ V,
    unsigned short* __restrict__ Ctx)   // (B,H,LQ, J*HD) bf16
{
    __shared__ unsigned short ldsVT[64 * 64];   // V^T tile: [feature][key]
    __shared__ unsigned short ldsP [64 * 64];   // probs (wave-private strips)

    const int lane = threadIdx.x & 31;
    const int wid  = threadIdx.x >> 5;
    const int r16  = lane & 15, half = lane >> 4;

    const int qt  = blockIdx.x;
    const int jbh = blockIdx.y;
    const int h = jbh % NH;
    const int b = (jbh / NH) % NB;
    const int j = jbh / (NH * NB);

    const unsigned short* Qh = Q  + ((size_t)b * LQn) * Dn + h * HDn;
    const unsigned short* Kh = Kk + ((size_t)(j * NB + b) * LKn) * Dn + h * HDn;
    const unsigned short* Vh = V  + ((size_t)(j * NB + b) * LKn) * Dn + h * HDn;

    const int m0 = qt * 64 + wid * 16;
    const v16u qf0 = ldfrag_bf(Qh + (size_t)m0 * Dn,      Dn, lane);
    const v16u qf1 = ldfrag_bf(Qh + (size_t)m0 * Dn + 32, Dn, lane);

    float rmax[8], rsum[8];
#pragma unroll
    for (int i = 0; i < 8; ++i) { rmax[i] = -1e30f; rsum[i] = 0.0f; }
    v8f cacc[4] = {};

    for (int t = 0; t < LKn / 64; ++t) {
        __syncthreads();                        // done reading previous V^T
        for (int idx = threadIdx.x; idx < 64 * 64; idx += 128) {
            const int kk = idx >> 6, nn = idx & 63;
            ldsVT[nn * 64 + kk] = Vh[(size_t)(t * 64 + kk) * Dn + nn];
        }

        // S = (Q K^T) / 8   (wave strip: 16 rows x 64 keys)
        v8f s[4] = {};
#pragma unroll
        for (int nt = 0; nt < 4; ++nt) {
            const unsigned short* kb = Kh + (size_t)(t * 64 + nt * 16) * Dn;
            v16u kf0 = ldfrag_bf(kb,      Dn, lane);
            v16u kf1 = ldfrag_bf(kb + 32, Dn, lane);
            s[nt] = wmma_bf16(qf0, kf0, s[nt]);
            s[nt] = wmma_bf16(qf1, kf1, s[nt]);
        }

        // per-row tile max (reduce over 16 lanes of each half-wave)
        float tmax[8];
#pragma unroll
        for (int i = 0; i < 8; ++i) {
#pragma unroll
            for (int nt = 0; nt < 4; ++nt) s[nt][i] *= 0.125f;
            float m = fmaxf(fmaxf(s[0][i], s[1][i]), fmaxf(s[2][i], s[3][i]));
#pragma unroll
            for (int d = 1; d < 16; d <<= 1) m = fmaxf(m, __shfl_xor(m, d, 32));
            tmax[i] = m;
        }
        // online rescale of running state
#pragma unroll
        for (int i = 0; i < 8; ++i) {
            const float nm = fmaxf(rmax[i], tmax[i]);
            const float corr = __expf(rmax[i] - nm);
            rmax[i] = nm;
            rsum[i] *= corr;
#pragma unroll
            for (int nt = 0; nt < 4; ++nt) cacc[nt][i] *= corr;
        }
        // P = exp(S - max); row sums; stash P (bf16) in wave-private LDS strip
#pragma unroll
        for (int i = 0; i < 8; ++i) {
            float ps = 0.0f;
#pragma unroll
            for (int nt = 0; nt < 4; ++nt) {
                const float p = __expf(s[nt][i] - rmax[i]);
                ps += p;
                ldsP[(wid * 16 + half * 8 + i) * 64 + nt * 16 + r16] = f2bf(p);
            }
#pragma unroll
            for (int d = 1; d < 16; d <<= 1) ps += __shfl_xor(ps, d, 32);
            rsum[i] += ps;
        }
        __syncthreads();                        // V^T tile ready

        // O += P @ V   (A = P strip from LDS, B = V^T rows from LDS)
#pragma unroll
        for (int ks = 0; ks < 2; ++ks) {
            v16u pf = ldfrag_bf(ldsP + (wid * 16) * 64 + ks * 32, 64, lane);
#pragma unroll
            for (int nt = 0; nt < 4; ++nt) {
                v16u vf = ldfrag_bf(ldsVT + (nt * 16) * 64 + ks * 32, 64, lane);
                cacc[nt] = wmma_bf16(pf, vf, cacc[nt]);
            }
        }
    }

    // ctx[b,h,q, j*64 + f] = O / l
#pragma unroll
    for (int i = 0; i < 8; ++i) {
        const float inv = 1.0f / rsum[i];
        const int q = qt * 64 + wid * 16 + half * 8 + i;
        const size_t ro = (((size_t)b * NH + h) * LQn + q) * (NJ * HDn) + (size_t)j * HDn;
#pragma unroll
        for (int nt = 0; nt < 4; ++nt)
            Ctx[ro + nt * 16 + r16] = f2bf(cacc[nt][i] * inv);
    }
}

// ==================================================================
// Kernel 3: fused per-head MLP: GELU(ctx@W1^T+b1)@W2^T+b2 -> (B,LQ,D) f32
// grid = (B*H*LQ)/64, block = 128 thr (4 waves); wave owns 16 rows.
// All operands pre-cast bf16.
// ==================================================================
__global__ __launch_bounds__(128) void mlp_kernel(
    const unsigned short* __restrict__ Ctx,   // (R, 192) bf16
    const unsigned short* __restrict__ W1, const float* __restrict__ b1,
    const unsigned short* __restrict__ W2, const float* __restrict__ b2,
    float* __restrict__ Out)
{
    __shared__ unsigned short ldsH[64 * 64];
    const int lane = threadIdx.x & 31;
    const int wid  = threadIdx.x >> 5;
    const int r16  = lane & 15, half = lane >> 4;
    const int row0 = blockIdx.x * 64 + wid * 16;
    const int KIN = NJ * HDn;                 // 192

    v8f acc1[4] = {};
#pragma unroll
    for (int ks = 0; ks < KIN / 32; ++ks) {
        v16u af = ldfrag_bf(Ctx + (size_t)row0 * KIN + ks * 32, KIN, lane);
#pragma unroll
        for (int nt = 0; nt < 4; ++nt) {
            v16u wf = ldfrag_bf(W1 + (size_t)(nt * 16) * KIN + ks * 32, KIN, lane);
            acc1[nt] = wmma_bf16(af, wf, acc1[nt]);
        }
    }
    // bias + exact GELU, stage as bf16 A-operand strip in LDS
#pragma unroll
    for (int nt = 0; nt < 4; ++nt) {
        const float bv = b1[nt * 16 + r16];
#pragma unroll
        for (int i = 0; i < 8; ++i) {
            const float x = acc1[nt][i] + bv;
            const float g = 0.5f * x * (1.0f + erff(x * 0.70710678118654752f));
            ldsH[(wid * 16 + half * 8 + i) * 64 + nt * 16 + r16] = f2bf(g);
        }
    }
    // second GEMM (wave-private LDS strip; same-wave DS ops are in-order)
    v8f acc2[4] = {};
#pragma unroll
    for (int ks = 0; ks < 2; ++ks) {
        v16u af = ldfrag_bf(ldsH + (wid * 16) * 64 + ks * 32, 64, lane);
#pragma unroll
        for (int nt = 0; nt < 4; ++nt) {
            v16u wf = ldfrag_bf(W2 + (size_t)(nt * 16) * 64 + ks * 32, 64, lane);
            acc2[nt] = wmma_bf16(af, wf, acc2[nt]);
        }
    }
    // scatter to (B, LQ, H*HD)
#pragma unroll
    for (int nt = 0; nt < 4; ++nt) {
        const float bv = b2[nt * 16 + r16];
#pragma unroll
        for (int i = 0; i < 8; ++i) {
            const int r  = row0 + half * 8 + i;
            const int bb = r / (NH * LQn);
            const int hh = (r / LQn) % NH;
            const int qq = r % LQn;
            Out[((size_t)bb * LQn + qq) * Dn + hh * HDn + nt * 16 + r16] = acc2[nt][i] + bv;
        }
    }
}

// ==================================================================
extern "C" void kernel_launch(void* const* d_in, const int* in_sizes, int n_in,
                              void* d_out, int out_size, void* d_ws, size_t ws_size,
                              hipStream_t stream) {
    (void)in_sizes; (void)n_in; (void)out_size; (void)ws_size;

    const float* q_in = (const float*)d_in[0];   // (B,LQ,D)
    const float* k_in = (const float*)d_in[1];   // (J,B,LK,D)
    const float* v_in = (const float*)d_in[2];
    const float* Wq = (const float*)d_in[3];  const float* bq = (const float*)d_in[4];
    const float* Wk = (const float*)d_in[5];  const float* bk = (const float*)d_in[6];
    const float* Wv = (const float*)d_in[7];  const float* bv = (const float*)d_in[8];
    const float* W1 = (const float*)d_in[9];  const float* b1 = (const float*)d_in[10];
    const float* W2 = (const float*)d_in[11]; const float* b2 = (const float*)d_in[12];
    float* out = (float*)d_out;

    // ---- workspace partition (bf16 = ushort); all region sizes are /8 ----
    const size_t nQ  = (size_t)NB * LQn * Dn;            //  8,388,608
    const size_t nKV = (size_t)NJ * NB * LKn * Dn;       // 25,165,824
    const size_t nW  = (size_t)Dn * Dn;                  //  1,048,576
    const size_t nW1 = (size_t)HDn * (NJ * HDn);         //     12,288
    const size_t nW2 = (size_t)HDn * HDn;                //      4,096

    unsigned short* xq  = (unsigned short*)d_ws;  // cast of query_states
    unsigned short* xk  = xq + nQ;                // cast of key_states
    unsigned short* xv  = xk + nKV;               // cast of value_states
    unsigned short* qb  = xv + nKV;               // q projection (bf16)
    unsigned short* kb  = qb + nQ;                // k projection
    unsigned short* vb  = kb + nKV;               // v projection
    unsigned short* wqb = vb + nKV;               // weight casts
    unsigned short* wkb = wqb + nW;
    unsigned short* wvb = wkb + nW;
    unsigned short* w1b = wvb + nW;
    unsigned short* w2b = w1b + nW1;
    unsigned short* ctx = xk;                     // alias: xk dead after k-proj GEMM

    auto cast = [&](const float* src, unsigned short* dst, size_t n) {
        const unsigned blocks = (unsigned)((n / 8 + 255) / 256);
        cast_f32_bf16_kernel<<<dim3(blocks), 256, 0, stream>>>(src, dst, n);
    };
    cast(q_in, xq, nQ);
    cast(k_in, xk, nKV);
    cast(v_in, xv, nKV);
    cast(Wq, wqb, nW);
    cast(Wk, wkb, nW);
    cast(Wv, wvb, nW);
    cast(W1, w1b, nW1);
    cast(W2, w2b, nW2);

    const int Mq  = NB * LQn;            // 8192
    const int Mkv = NJ * NB * LKn;       // 24576

    proj_gemm_kernel<<<dim3(Dn / 128, Mq  / 128), 256, 0, stream>>>(xq, wqb, bq, qb, Mq,  Dn, Dn);
    proj_gemm_kernel<<<dim3(Dn / 128, Mkv / 128), 256, 0, stream>>>(xk, wkb, bk, kb, Mkv, Dn, Dn);
    proj_gemm_kernel<<<dim3(Dn / 128, Mkv / 128), 256, 0, stream>>>(xv, wvb, bv, vb, Mkv, Dn, Dn);

    attn_kernel<<<dim3(LQn / 64, NJ * NB * NH), 128, 0, stream>>>(qb, kb, vb, ctx);

    mlp_kernel<<<dim3((NB * NH * LQn) / 64), 128, 0, stream>>>(ctx, w1b, b1, w2b, b2, out);
}
